// CIDER_25383256719526
// MI455X (gfx1250) — matrix operand order repaired and checked by reading, not compiled, not run on hardware
//
#include <hip/hip_runtime.h>
#include <hip/hip_bf16.h>

typedef __attribute__((ext_vector_type(16))) __bf16 v16bf;
typedef __attribute__((ext_vector_type(8)))  __bf16 v8bf;
typedef __attribute__((ext_vector_type(8)))  float  v8f;

static constexpr int BB = 96;    // batch
static constexpr int HH = 100;   // history length
static constexpr int NCAND = 128;
static constexpr int DD = 768;   // embed dim
static constexpr int AA = 256;   // attention dim

// ---------- helpers ----------
__device__ __forceinline__ __bf16 f2bf(float f) {
  unsigned u = __float_as_uint(f);
  u += 0x7fffu + ((u >> 16) & 1u);          // round-to-nearest-even
  unsigned short h = (unsigned short)(u >> 16);
  __bf16 r;
  __builtin_memcpy(&r, &h, sizeof(r));
  return r;
}

// ---------- elementwise f32 -> bf16 ----------
__global__ void k_f2bf(const float* __restrict__ x, __bf16* __restrict__ y, int n) {
  int i = blockIdx.x * blockDim.x + threadIdx.x;
  if (i < n) y[i] = f2bf(x[i]);
}

// ---------- neigh[b,d] = mean_{h<96} hist[b,h,d]  (bf16 out) ----------
__global__ void k_neigh(const float* __restrict__ hist, __bf16* __restrict__ neigh) {
  int i = blockIdx.x * blockDim.x + threadIdx.x;
  if (i >= BB * DD) return;
  int b = i / DD, d = i % DD;
  const float* p = hist + (size_t)b * HH * DD + d;
  float s = 0.f;
#pragma unroll 4
  for (int h = 0; h < BB; ++h) s += p[(size_t)h * DD];
  neigh[i] = f2bf(s * (1.0f / BB));
}

// ---------- gather user_rep rows (gcn[:,0,:]) into compact bf16 ----------
__global__ void k_gather_urep(const __bf16* __restrict__ gcn_bf, __bf16* __restrict__ urep) {
  int i = blockIdx.x * blockDim.x + threadIdx.x;
  if (i >= BB * DD) return;
  int b = i / DD, d = i % DD;
  urep[i] = gcn_bf[(size_t)b * HH * DD + d];
}

// ---------- generic WMMA GEMM:  C[m,n] = sum_k A[m,k]*W[n,k] (+ biases) ----------
// A: M x K bf16 row-major, W: N x K bf16 row-major, C: M x N f32.
// Each wave computes a 16x64 tile: one A fragment reused across 4 WMMAs.
// Requires M%16==0, N%64==0, K%32==0.  EXEC stays all-ones around WMMA.
__global__ void k_wmma_gemm_nt(const __bf16* __restrict__ A,
                               const __bf16* __restrict__ W,
                               float* __restrict__ C,
                               __bf16* __restrict__ Cbf,          // optional bf16 mirror
                               const float* __restrict__ bias_n,  // [N] or null
                               const float* __restrict__ bias_bn, // [M/hdiv, N] or null
                               int hdiv, int M, int N, int K) {
  int wave = (blockIdx.x * blockDim.x + threadIdx.x) >> 5;
  int ngrp = N >> 6;                // 64-column groups
  int mt = wave / ngrp;
  int ng = wave % ngrp;
  if (mt * 16 >= M) return;         // wave-uniform

  int lane = threadIdx.x & 31;
  int half = lane >> 4;             // lane group 0..15 vs 16..31
  int r    = lane & 15;
  int m0 = mt * 16;
  int n0 = ng * 64;

  const __bf16* arow = A + (size_t)(m0 + r) * K;
  v8f acc0 = {}, acc1 = {}, acc2 = {}, acc3 = {};

  for (int k = 0; k < K; k += 32) {
    // A fragment (16x32): lane holds row r, K = half*16+[0..15]
    v8bf alo = *(const v8bf*)(arow + k + half * 16);
    v8bf ahi = *(const v8bf*)(arow + k + half * 16 + 8);
    v16bf af = __builtin_shufflevector(alo, ahi,
        0, 1, 2, 3, 4, 5, 6, 7, 8, 9, 10, 11, 12, 13, 14, 15);
    // B fragments (32x16): lane holds column n0+j*16+r, K = half*16+[0..15] (contig 32B)
    const __bf16* wbase = W + (size_t)(n0 + r) * K + k + half * 16;
    v16bf b0 = *(const v16bf*)(wbase);
    v16bf b1 = *(const v16bf*)(wbase + (size_t)16 * K);
    v16bf b2 = *(const v16bf*)(wbase + (size_t)32 * K);
    v16bf b3 = *(const v16bf*)(wbase + (size_t)48 * K);
    acc0 = __builtin_amdgcn_wmma_f32_16x16x32_bf16(false, af, false, b0, (short)0, acc0, false, false);
    acc1 = __builtin_amdgcn_wmma_f32_16x16x32_bf16(false, af, false, b1, (short)0, acc1, false, false);
    acc2 = __builtin_amdgcn_wmma_f32_16x16x32_bf16(false, af, false, b2, (short)0, acc2, false, false);
    acc3 = __builtin_amdgcn_wmma_f32_16x16x32_bf16(false, af, false, b3, (short)0, acc3, false, false);
  }

  v8f accs[4] = {acc0, acc1, acc2, acc3};
#pragma unroll
  for (int j = 0; j < 4; ++j) {
#pragma unroll
    for (int v = 0; v < 8; ++v) {
      int m = m0 + half * 8 + v;     // C layout: VGPR v <-> row v + 8*half
      int n = n0 + j * 16 + r;       // lane r <-> column
      float val = accs[j][v];
      if (bias_n)  val += bias_n[n];
      if (bias_bn) val += bias_bn[(size_t)(m / hdiv) * N + n];
      C[(size_t)m * N + n] = val;
      if (Cbf) Cbf[(size_t)m * N + n] = f2bf(val);
    }
  }
}

// ---------- fused scores -> softmax -> weighted sum -> broadcast ----------
// one block per batch element b, 256 threads (8 waves)
__global__ void k_attn_out(const float* __restrict__ Kmat,  // (B*H, A)
                           const float* __restrict__ Q,     // (B, A)
                           const float* __restrict__ gcn,   // (B*H, D) f32
                           float* __restrict__ out) {       // (B, NCAND, D)
  int b = blockIdx.x;
  int t = threadIdx.x;
  __shared__ float sQ[AA];
  __shared__ float sS[HH];

  sQ[t] = Q[(size_t)b * AA + t];     // blockDim == AA == 256
  __syncthreads();

  int wave = t >> 5, lane = t & 31;
  for (int h = wave; h < HH; h += 8) {
    const float* kr = Kmat + (size_t)(b * HH + h) * AA;
    float p = 0.f;
#pragma unroll
    for (int i = 0; i < AA / 32; ++i) p += kr[lane + i * 32] * sQ[lane + i * 32];
    for (int off = 16; off > 0; off >>= 1) p += __shfl_down(p, off, 32);
    if (lane == 0) sS[h] = p * 0.0625f;   // 1/sqrt(256)
  }
  __syncthreads();

  if (t == 0) {
    float mx = -3.4e38f;
    for (int h = 0; h < HH; ++h) mx = fmaxf(mx, sS[h]);
    float sum = 0.f;
    for (int h = 0; h < HH; ++h) { float e = expf(sS[h] - mx); sS[h] = e; sum += e; }
    float inv = 1.f / sum;
    for (int h = 0; h < HH; ++h) sS[h] *= inv;
  }
  __syncthreads();

  for (int d = t; d < DD; d += 256) {
    float acc = 0.f;
    const float* g = gcn + (size_t)b * HH * DD + d;
    for (int h = 0; h < HH; ++h) acc += sS[h] * g[(size_t)h * DD];
    float* ob = out + (size_t)b * NCAND * DD + d;
    for (int n = 0; n < NCAND; ++n) ob[(size_t)n * DD] = acc;  // broadcast over candidates
  }
}

// ---------- host ----------
extern "C" void kernel_launch(void* const* d_in, const int* in_sizes, int n_in,
                              void* d_out, int out_size, void* d_ws, size_t ws_size,
                              hipStream_t stream) {
  const float* hist = (const float*)d_in[0];
  // d_in[1] = candidate_news_representation (only its shape matters; unused)
  const float* W_l = (const float*)d_in[2];
  const float* b_l = (const float*)d_in[3];
  const float* W_r = (const float*)d_in[4];
  const float* W_K = (const float*)d_in[5];
  const float* W_Q = (const float*)d_in[6];
  const float* b_Q = (const float*)d_in[7];
  float* out = (float*)d_out;

  char* ws = (char*)d_ws;
  size_t off = 0;
  auto alloc = [&](size_t bytes) -> char* {
    char* p = ws + off;
    off = (off + bytes + 255) & ~(size_t)255;
    return p;
  };
  __bf16* hist_bf  = (__bf16*)alloc((size_t)BB * HH * DD * 2);
  __bf16* Wr_bf    = (__bf16*)alloc((size_t)DD * DD * 2);
  __bf16* Wl_bf    = (__bf16*)alloc((size_t)DD * DD * 2);
  __bf16* WK_bf    = (__bf16*)alloc((size_t)AA * DD * 2);
  __bf16* WQ_bf    = (__bf16*)alloc((size_t)AA * DD * 2);
  __bf16* neigh_bf = (__bf16*)alloc((size_t)BB * DD * 2);
  float*  bias_f   = (float*)alloc((size_t)BB * DD * 4);
  float*  gcn_f    = (float*)alloc((size_t)BB * HH * DD * 4);
  __bf16* gcn_bf   = (__bf16*)alloc((size_t)BB * HH * DD * 2);
  __bf16* urep_bf  = (__bf16*)alloc((size_t)BB * DD * 2);
  float*  Q_f      = (float*)alloc((size_t)BB * AA * 4);
  float*  Kmat_f   = (float*)alloc((size_t)BB * HH * AA * 4);
  (void)ws_size; (void)in_sizes; (void)n_in; (void)out_size;

  const int thr = 256;
  // 1) bf16 conversions
  k_f2bf<<<(BB * HH * DD + thr - 1) / thr, thr, 0, stream>>>(hist, hist_bf, BB * HH * DD);
  k_f2bf<<<(DD * DD + thr - 1) / thr, thr, 0, stream>>>(W_r, Wr_bf, DD * DD);
  k_f2bf<<<(DD * DD + thr - 1) / thr, thr, 0, stream>>>(W_l, Wl_bf, DD * DD);
  k_f2bf<<<(AA * DD + thr - 1) / thr, thr, 0, stream>>>(W_K, WK_bf, AA * DD);
  k_f2bf<<<(AA * DD + thr - 1) / thr, thr, 0, stream>>>(W_Q, WQ_bf, AA * DD);
  // 2) neighbor mean (over first 96 history rows, per reference's [:, :b, :])
  k_neigh<<<(BB * DD + thr - 1) / thr, thr, 0, stream>>>(hist, neigh_bf);
  // 3) bias[b,e] = neigh @ W_l^T + b_l   (M=96, N=768, K=768 -> 72 waves)
  k_wmma_gemm_nt<<<9, thr, 0, stream>>>(neigh_bf, Wl_bf, bias_f, nullptr,
                                        b_l, nullptr, 1, BB, DD, DD);
  // 4) gcn = hist @ W_r^T + bias[b,:]    (M=9600, N=768, K=768 -> 7200 waves)
  k_wmma_gemm_nt<<<900, thr, 0, stream>>>(hist_bf, Wr_bf, gcn_f, gcn_bf,
                                          nullptr, bias_f, HH, BB * HH, DD, DD);
  // 5) user_rep rows
  k_gather_urep<<<(BB * DD + thr - 1) / thr, thr, 0, stream>>>(gcn_bf, urep_bf);
  // 6) Q = user_rep @ W_Q^T + b_Q        (M=96, N=256, K=768 -> 24 waves)
  k_wmma_gemm_nt<<<3, thr, 0, stream>>>(urep_bf, WQ_bf, Q_f, nullptr,
                                        b_Q, nullptr, 1, BB, AA, DD);
  // 7) K = gcn @ W_K^T                   (M=9600, N=256, K=768 -> 2400 waves)
  k_wmma_gemm_nt<<<300, thr, 0, stream>>>(gcn_bf, WK_bf, Kmat_f, nullptr,
                                          nullptr, nullptr, 1, BB * HH, AA, DD);
  // 8) scores -> softmax -> weighted sum -> broadcast to (B, N, D)
  k_attn_out<<<BB, thr, 0, stream>>>(Kmat_f, Q_f, gcn_f, out);
}
